// SelfAttn_8289286881778
// MI455X (gfx1250) — matrix operand order repaired and checked by reading, compiled, not run on hardware
//
#include <hip/hip_runtime.h>

// ---------------------------------------------------------------------------
// MI455X (gfx1250) mixed-precision self-attention pipeline, round 2.
//
// ~161 GFLOP of GEMM vs ~0.5 GB of traffic -> matrix-core bound. All GEMMs run
// on v_wmma_f32_16x16x32_bf16 (f32 accumulate). The two large GEMMs (QKV and
// final projection, ~96% of FLOPs) now stage 128x128 block tiles through LDS
// using gfx1250 async-to-LDS copies (global_load_async_to_lds_b128, ASYNCcnt)
// with double buffering, so HBM latency overlaps WMMA compute and each global
// byte is loaded once per block instead of once per wave.
// ---------------------------------------------------------------------------

typedef __bf16 bf16_t;
typedef __attribute__((ext_vector_type(16))) __bf16 v16bf;
typedef __attribute__((ext_vector_type(8)))  __bf16 v8bf;
typedef __attribute__((ext_vector_type(8)))  float  v8f;

#define BQ      8
#define NTOK    4096
#define INDIM   768
#define NHEAD   12
#define DHEAD   64
#define DMODEL  768          // NHEAD * DHEAD
#define QKVDIM  2304         // 3 * DMODEL
#define SCALE   0.125f       // DHEAD^-0.5

// ---------------- CDNA5 async-copy helpers (inline asm, ASYNCcnt) -----------

__device__ __forceinline__ unsigned lds_off(const void* p) {
  // generic pointer to LDS: low 32 bits are the LDS byte offset
  return (unsigned)(uintptr_t)p;
}

__device__ __forceinline__ void async_b128(unsigned lds_addr, const void* gaddr) {
  // per-lane 16B copy global -> LDS, tracked with ASYNCcnt
  asm volatile("global_load_async_to_lds_b128 %0, %1, off"
               :: "v"(lds_addr), "v"(gaddr) : "memory");
}

__device__ __forceinline__ void wait_async0() {
  asm volatile("s_wait_asynccnt 0x0" ::: "memory");
}

// ---------------- WMMA fragment helpers (wave32 layouts, CDNA5 ISA 7.12.2) --

// Global A fragment 16x32 bf16: lane L<16 holds row M=L, K = kb+{0..7},{16..23};
// lanes 16..31 same rows, K offset +8.  Two 16B loads per lane.
__device__ __forceinline__ v16bf frag_a(const bf16_t* __restrict__ A, size_t lda,
                                        int m_base, int kb, int lane) {
  const int m  = lane & 15;
  const int hi = lane >> 4;
  const bf16_t* p = A + (size_t)(m_base + m) * lda + (size_t)kb + hi * 8;
  v8bf lo = *(const v8bf*)(p);
  v8bf hv = *(const v8bf*)(p + 16);
  v16bf a;
#pragma unroll
  for (int j = 0; j < 8; ++j) { a[j] = lo[j]; a[j + 8] = hv[j]; }
  return a;
}

// Global B fragment 32x16 bf16, B stored [Ncols][K]: lane L<16 holds column
// N=L, K = kb+{0..7},{16..23}; lanes 16..31 same columns, K offset +8.
__device__ __forceinline__ v16bf frag_b(const bf16_t* __restrict__ Bm, size_t ldb,
                                        int n_base, int kb, int lane) {
  const int n  = lane & 15;
  const int hi = lane >> 4;
  const bf16_t* p = Bm + (size_t)(n_base + n) * ldb + (size_t)kb + hi * 8;
  v8bf lo = *(const v8bf*)(p);
  v8bf hv = *(const v8bf*)(p + 16);
  v16bf b;
#pragma unroll
  for (int j = 0; j < 8; ++j) { b[j] = lo[j]; b[j + 8] = hv[j]; }
  return b;
}

__device__ __forceinline__ v8f wmma_bf16(v16bf a, v16bf b, v8f c) {
  return __builtin_amdgcn_wmma_f32_16x16x32_bf16(false, a, false, b,
                                                 (short)0, c, false, false);
}

// ---------------- LDS fragment loads (rows padded to 80B for bank spread) ---

#define BM 128
#define BN 128
#define BK 32
#define LDS_ROW 80                    // 64B of K data + 16B pad
#define TILE_BYTES (128 * LDS_ROW)    // 10240 B per tile

__device__ __forceinline__ v16bf lds_frag_a(const char* tile, int row0, int lane) {
  const int m  = row0 + (lane & 15);
  const int hi = lane >> 4;
  const char* p = tile + m * LDS_ROW + hi * 16;
  v8bf lo = *(const v8bf*)(p);
  v8bf hv = *(const v8bf*)(p + 32);
  v16bf a;
#pragma unroll
  for (int j = 0; j < 8; ++j) { a[j] = lo[j]; a[j + 8] = hv[j]; }
  return a;
}

__device__ __forceinline__ v16bf lds_frag_b(const char* tile, int row0, int lane) {
  const int nn = row0 + (lane & 15);
  const int hi = lane >> 4;
  const char* p = tile + nn * LDS_ROW + hi * 16;
  v8bf lo = *(const v8bf*)(p);
  v8bf hv = *(const v8bf*)(p + 32);
  v16bf b;
#pragma unroll
  for (int j = 0; j < 8; ++j) { b[j] = lo[j]; b[j + 8] = hv[j]; }
  return b;
}

// ---------------- big GEMM: C = A * B^T (+bias), LDS-staged, double-buffered
// Block: 256 threads = 8 waves computes 128x128 C tile. Wave (4Mx2N grid)
// computes 32x64 = 8 WMMA tiles. K consumed in 32-deep chunks.
__global__ __launch_bounds__(256) void gemm_lds_bf16_f32(
    const bf16_t* __restrict__ A, const bf16_t* __restrict__ Bm,
    float* __restrict__ C, const float* __restrict__ bias,
    int K, int lda, int ldb, int ldc) {
  __shared__ __align__(16) char smem[2][2 * TILE_BYTES];  // 40 KB

  const int t    = threadIdx.x;
  const int lane = t & 31;
  const int wave = t >> 5;
  const int m_off = (wave & 3) * 32;
  const int n_off = (wave >> 2) * 64;
  const size_t blockM = (size_t)blockIdx.y * BM;
  const size_t blockN = (size_t)blockIdx.x * BN;

  // async-copy mapping: 512 x 16B per tile, 2 per thread per tile
  const int seg  = t & 3;        // 16B segment within a 64B K-row
  const int row0 = t >> 2;       // rows row0 and row0+64
  const char* Ag = (const char*)(A + blockM * (size_t)lda);
  const char* Bg = (const char*)(Bm + blockN * (size_t)ldb);
  const size_t ldab = (size_t)lda * 2, ldbb = (size_t)ldb * 2;

  auto load_tile = [&](int buf, int kb) {
    char* As = smem[buf];
    char* Bs = smem[buf] + TILE_BYTES;
    const size_t kbyte = (size_t)kb * 2 + seg * 16;
    async_b128(lds_off(As + row0 * LDS_ROW + seg * 16),
               Ag + (size_t)row0 * ldab + kbyte);
    async_b128(lds_off(As + (row0 + 64) * LDS_ROW + seg * 16),
               Ag + (size_t)(row0 + 64) * ldab + kbyte);
    async_b128(lds_off(Bs + row0 * LDS_ROW + seg * 16),
               Bg + (size_t)row0 * ldbb + kbyte);
    async_b128(lds_off(Bs + (row0 + 64) * LDS_ROW + seg * 16),
               Bg + (size_t)(row0 + 64) * ldbb + kbyte);
  };

  v8f acc[2][4] = {};

  load_tile(0, 0);
  wait_async0();
  __syncthreads();

  const int nk = K / BK;
  for (int c = 0; c < nk; ++c) {
    const int cur = c & 1;
    if (c + 1 < nk) load_tile(cur ^ 1, (c + 1) * BK);   // prefetch next chunk

    const char* As = smem[cur];
    const char* Bs = smem[cur] + TILE_BYTES;
    v16bf a0 = lds_frag_a(As, m_off, lane);
    v16bf a1 = lds_frag_a(As, m_off + 16, lane);
#pragma unroll
    for (int tn = 0; tn < 4; ++tn) {
      v16bf b = lds_frag_b(Bs, n_off + tn * 16, lane);
      acc[0][tn] = wmma_bf16(a0, b, acc[0][tn]);
      acc[1][tn] = wmma_bf16(a1, b, acc[1][tn]);
    }

    wait_async0();      // our async writes for next buffer are done
    __syncthreads();    // everyone finished reading cur, writes published
  }

  const int n = lane & 15, hi = lane >> 4;
#pragma unroll
  for (int sm = 0; sm < 2; ++sm) {
#pragma unroll
    for (int tn = 0; tn < 4; ++tn) {
      const size_t col = blockN + n_off + tn * 16 + n;
      const float badd = bias ? bias[col] : 0.0f;
#pragma unroll
      for (int r = 0; r < 8; ++r) {
        const size_t row = blockM + m_off + sm * 16 + r + (hi << 3);
        C[row * (size_t)ldc + col] = acc[sm][tn][r] + badd;
      }
    }
  }
}

// ---------------- elementwise f32 -> bf16 -----------------------------------
__global__ void f32_to_bf16(const float* __restrict__ in, bf16_t* __restrict__ out,
                            size_t n) {
  size_t i = (size_t)blockIdx.x * blockDim.x + threadIdx.x;
  if (i < n) out[i] = (bf16_t)in[i];
}

// ---------------- q softmax over dh=64, * SCALE, -> bf16 [bh][n][d] ---------
__global__ void softmax_q(const float* __restrict__ qkv, bf16_t* __restrict__ qout) {
  const int lane = threadIdx.x & 31;
  const int row  = blockIdx.x * (blockDim.x >> 5) + (threadIdx.x >> 5); // (b*H+h)*N + n
  const int n  = row & (NTOK - 1);
  const int bh = row >> 12;
  const int h  = bh % NHEAD;
  const int b  = bh / NHEAD;
  const float* src = qkv + ((size_t)(b * NTOK + n)) * QKVDIM + h * 192;
  float v0 = src[lane], v1 = src[lane + 32];
  float m = fmaxf(v0, v1);
#pragma unroll
  for (int off = 16; off; off >>= 1) m = fmaxf(m, __shfl_xor(m, off, 32));
  float e0 = __expf(v0 - m), e1 = __expf(v1 - m);
  float s = e0 + e1;
#pragma unroll
  for (int off = 16; off; off >>= 1) s += __shfl_xor(s, off, 32);
  const float inv = SCALE / s;
  bf16_t* dst = qout + (size_t)row * DHEAD;
  dst[lane]      = (bf16_t)(e0 * inv);
  dst[lane + 32] = (bf16_t)(e1 * inv);
}

// ---------------- k softmax over tokens, transposed store [bh][d][n] --------
__global__ void softmax_k_T(const float* __restrict__ qkv, bf16_t* __restrict__ kT) {
  __shared__ float red[256];
  const int col = blockIdx.x;           // (b*H+h)*64 + d
  const int d  = col & 63;
  const int bh = col >> 6;
  const int h  = bh % NHEAD;
  const int b  = bh / NHEAD;
  const float* base = qkv + (size_t)b * NTOK * QKVDIM + h * 192 + DHEAD + d;
  const int t = threadIdx.x;

  float m = -3.4e38f;
  for (int n = t; n < NTOK; n += 256) m = fmaxf(m, base[(size_t)n * QKVDIM]);
  red[t] = m; __syncthreads();
  for (int s = 128; s; s >>= 1) { if (t < s) red[t] = fmaxf(red[t], red[t + s]); __syncthreads(); }
  m = red[0]; __syncthreads();

  float sum = 0.0f;
  for (int n = t; n < NTOK; n += 256) sum += __expf(base[(size_t)n * QKVDIM] - m);
  red[t] = sum; __syncthreads();
  for (int s = 128; s; s >>= 1) { if (t < s) red[t] += red[t + s]; __syncthreads(); }
  const float inv = 1.0f / red[0];

  bf16_t* dst = kT + (size_t)col * NTOK;
  for (int n = t; n < NTOK; n += 256)
    dst[n] = (bf16_t)(__expf(base[(size_t)n * QKVDIM] - m) * inv);
}

// ---------------- v transpose -> bf16 [bh][e][n] ----------------------------
__global__ void transpose_v(const float* __restrict__ qkv, bf16_t* __restrict__ vT) {
  size_t i = (size_t)blockIdx.x * blockDim.x + threadIdx.x;  // (bh,e,n) linear
  if (i >= (size_t)BQ * NHEAD * DHEAD * NTOK) return;
  const int n  = (int)(i & (NTOK - 1));
  const size_t r = i >> 12;
  const int e  = (int)(r & 63);
  const int bh = (int)(r >> 6);
  const int h  = bh % NHEAD;
  const int b  = bh / NHEAD;
  vT[i] = (bf16_t)qkv[((size_t)(b * NTOK + n)) * QKVDIM + h * 192 + 128 + e];
}

// ---------------- context[d][e] = sum_n kT[d][n]*vT[e][n]; store ctxT[e][d] -
__global__ void context_gemm(const bf16_t* __restrict__ kT,
                             const bf16_t* __restrict__ vT,
                             bf16_t* __restrict__ ctxT) {
  const int lane = threadIdx.x & 31;
  const int wave = blockIdx.x * (blockDim.x >> 5) + (threadIdx.x >> 5);
  const int bh = wave >> 2;           // 96 heads
  const int sm = wave & 3;            // 4 supertiles of 16 rows (d)
  const bf16_t* Ak = kT + (size_t)bh * DHEAD * NTOK;
  const bf16_t* Bv = vT + (size_t)bh * DHEAD * NTOK;

  v8f acc[4] = {};
  for (int kb = 0; kb < NTOK; kb += 32) {
    v16bf a = frag_a(Ak, NTOK, sm << 4, kb, lane);
#pragma unroll
    for (int t = 0; t < 4; ++t)
      acc[t] = wmma_bf16(a, frag_b(Bv, NTOK, t * 16, kb, lane), acc[t]);
  }

  const int n = lane & 15, hi = lane >> 4;
  bf16_t* out = ctxT + (size_t)bh * DHEAD * DHEAD;
#pragma unroll
  for (int t = 0; t < 4; ++t) {
    const int e = t * 16 + n;
#pragma unroll
    for (int r = 0; r < 8; ++r) {
      const int d = (sm << 4) + r + (hi << 3);
      out[(size_t)e * DHEAD + d] = (bf16_t)acc[t][r];  // transposed store
    }
  }
}

// ---------------- o = q @ context, merged-head bf16 store [b][n][h*64+e] ----
__global__ void o_gemm(const bf16_t* __restrict__ q,
                       const bf16_t* __restrict__ ctxT,
                       bf16_t* __restrict__ o) {
  const int lane = threadIdx.x & 31;
  const int wave = blockIdx.x * (blockDim.x >> 5) + (threadIdx.x >> 5);
  const int bh = wave >> 8;           // 256 M-supertiles per head
  const int sm = wave & 255;
  const int h = bh % NHEAD;
  const int b = bh / NHEAD;
  const bf16_t* A  = q    + (size_t)bh * NTOK * DHEAD;
  const bf16_t* Bm = ctxT + (size_t)bh * DHEAD * DHEAD;

  v8f acc[4] = {};
#pragma unroll
  for (int kb = 0; kb < DHEAD; kb += 32) {
    v16bf a = frag_a(A, DHEAD, sm << 4, kb, lane);
#pragma unroll
    for (int t = 0; t < 4; ++t)
      acc[t] = wmma_bf16(a, frag_b(Bm, DHEAD, t * 16, kb, lane), acc[t]);
  }

  const int n = lane & 15, hi = lane >> 4;
#pragma unroll
  for (int t = 0; t < 4; ++t) {
    const int e = t * 16 + n;
#pragma unroll
    for (int r = 0; r < 8; ++r) {
      const int row = (sm << 4) + r + (hi << 3);
      o[((size_t)(b * NTOK + row)) * DMODEL + h * DHEAD + e] = (bf16_t)acc[t][r];
    }
  }
}

// ---------------------------------------------------------------------------
extern "C" void kernel_launch(void* const* d_in, const int* in_sizes, int n_in,
                              void* d_out, int out_size, void* d_ws, size_t ws_size,
                              hipStream_t stream) {
  const float* x     = (const float*)d_in[0];   // [8,4096,768]
  const float* w_qkv = (const float*)d_in[1];   // [2304,768]
  const float* w_lin = (const float*)d_in[2];   // [768,768]
  const float* b_lin = (const float*)d_in[3];   // [768]
  float* out = (float*)d_out;                   // [8,4096,768]

  const int M = BQ * NTOK;                      // 32768 token rows

  // workspace partition (256B aligned)
  char* ws = (char*)d_ws;
  size_t off = 0;
  auto take = [&](size_t bytes) { char* p = ws + off; off = (off + bytes + 255) & ~(size_t)255; return p; };
  bf16_t* x_bf   = (bf16_t*)take((size_t)M * INDIM * 2);                 // 50 MB
  bf16_t* wq_bf  = (bf16_t*)take((size_t)QKVDIM * INDIM * 2);            // 3.5 MB
  bf16_t* wl_bf  = (bf16_t*)take((size_t)INDIM * DMODEL * 2);            // 1.2 MB
  float*  qkv    = (float*) take((size_t)M * QKVDIM * 4);                // 302 MB
  bf16_t* q_bf   = (bf16_t*)take((size_t)BQ * NHEAD * NTOK * DHEAD * 2); // 50 MB
  bf16_t* kT     = (bf16_t*)take((size_t)BQ * NHEAD * DHEAD * NTOK * 2); // 50 MB
  bf16_t* vT     = (bf16_t*)take((size_t)BQ * NHEAD * DHEAD * NTOK * 2); // 50 MB
  bf16_t* ctxT   = (bf16_t*)take((size_t)BQ * NHEAD * DHEAD * DHEAD * 2);
  bf16_t* o_bf   = (bf16_t*)take((size_t)M * DMODEL * 2);                // 50 MB
  (void)ws_size; (void)n_in; (void)in_sizes; (void)out_size;

  // 1) cast inputs to bf16 once
  {
    size_t nx = (size_t)M * INDIM;
    f32_to_bf16<<<(int)((nx + 255) / 256), 256, 0, stream>>>(x, x_bf, nx);
    size_t nw = (size_t)QKVDIM * INDIM;
    f32_to_bf16<<<(int)((nw + 255) / 256), 256, 0, stream>>>(w_qkv, wq_bf, nw);
    size_t nl = (size_t)INDIM * DMODEL;
    f32_to_bf16<<<(int)((nl + 255) / 256), 256, 0, stream>>>(w_lin, wl_bf, nl);
  }

  // 2) QKV projection: qkv = x @ w_qkv^T   (32768 x 2304, K=768), LDS-staged
  {
    dim3 grid(QKVDIM / BN, M / BM);             // 18 x 256
    gemm_lds_bf16_f32<<<grid, 256, 0, stream>>>(x_bf, wq_bf, qkv, nullptr,
                                                INDIM, INDIM, INDIM, QKVDIM);
  }

  // 3) softmax over dh for q (row softmax, * dh^-0.5), bf16 [bh][n][d]
  {
    int rows = BQ * NHEAD * NTOK;               // 393216, 8 rows/block
    softmax_q<<<rows / 8, 256, 0, stream>>>(qkv, q_bf);
  }

  // 4) softmax over tokens for k, transposed bf16 store [bh][d][n]
  softmax_k_T<<<BQ * NHEAD * DHEAD, 256, 0, stream>>>(qkv, kT);

  // 5) v transpose -> bf16 [bh][e][n]
  {
    size_t nv = (size_t)BQ * NHEAD * DHEAD * NTOK;
    transpose_v<<<(int)((nv + 255) / 256), 256, 0, stream>>>(qkv, vT);
  }

  // 6) context = k^T v per head (64x64, K=4096), transposed bf16 store
  context_gemm<<<(BQ * NHEAD * 4) / 8, 256, 0, stream>>>(kT, vT, ctxT);

  // 7) o = q @ context per head, merged heads -> bf16 [b][n][768]
  o_gemm<<<(BQ * NHEAD * 256) / 8, 256, 0, stream>>>(q_bf, ctxT, o_bf);

  // 8) final projection: out = o @ w_lin^T + b_lin (32768 x 768, K=768), LDS-staged
  {
    dim3 grid(DMODEL / BN, M / BM);             // 6 x 256
    gemm_lds_bf16_f32<<<grid, 256, 0, stream>>>(o_bf, wl_bf, out, b_lin,
                                                INDIM, DMODEL, DMODEL, DMODEL);
  }
}
